// Attention_80410377716310
// MI455X (gfx1250) — compile-verified, matrix operand order checked by reading, not run on hardware
//
#include <hip/hip_runtime.h>
#include <hip/hip_bf16.h>

typedef __attribute__((ext_vector_type(16))) _Float16 v16h;
typedef __attribute__((ext_vector_type(8)))  _Float16 v8h;
typedef __attribute__((ext_vector_type(8)))  float    v8f;

#define SCALE_F 0.125f     // HEAD_DIM=64 -> 1/sqrt(64)
#define NEG_BIG (-3.0e38f)

__device__ __forceinline__ v8f wmma16(v16h a, v16h b, v8f c) {
  return __builtin_amdgcn_wmma_f32_16x16x32_f16(false, a, false, b, (short)0, c, false, false);
}

// Per-lane async copy of 16 bytes global -> LDS (tracked on ASYNCcnt).
// lds_off = low 32 bits of a generic shared pointer (= LDS byte offset).
__device__ __forceinline__ void async_cp_b128(const _Float16* gsrc, void* lds_dst) {
  unsigned lds_off = (unsigned)(unsigned long long)(uintptr_t)lds_dst;
  unsigned long long gaddr = (unsigned long long)(uintptr_t)gsrc;
  asm volatile("global_load_async_to_lds_b128 %0, %1, off"
               :: "v"(lds_off), "v"(gaddr) : "memory");
}

__device__ __forceinline__ void wait_async0() {
  asm volatile("s_wait_asynccnt 0" ::: "memory");
}

// Load a 16x32 f16 fragment (A layout, or B layout when source is staged
// transposed [n][k]) from LDS. Per the CDNA5 ISA layout, lane (hi,m) holds
// K = {8*hi .. 8*hi+7} and {16+8*hi .. 23+8*hi} of row m -> two b128 loads.
__device__ __forceinline__ v16h frag_ld(const _Float16* base, int ld) {
  const int lane = threadIdx.x & 31;
  const int m = lane & 15;
  const int hi = lane >> 4;
  const _Float16* row = base + m * ld;
  v8h lo = *(const v8h*)(row + 8 * hi);
  v8h hh = *(const v8h*)(row + 16 + 8 * hi);
  v16h r;
#pragma unroll
  for (int i = 0; i < 8; ++i) { r[i] = lo[i]; r[i + 8] = hh[i]; }
  return r;
}

__device__ __forceinline__ v8f vzero8() {
  v8f z = {0.f, 0.f, 0.f, 0.f, 0.f, 0.f, 0.f, 0.f};
  return z;
}

// ---------------------------------------------------------------------------
// Kernel 1: qkv = x(f32)[32768x512] @ w_qkv(f32)[512x1536], emitted as f16
// into ws laid out [which(3)][B(32)][head(8)][n(1024)][d(64)].
// Block = 256 thr (8 waves), block tile 128x64, wave tile 32x32 (2x2 WMMA).
// f32 inputs must pass through VALU for f16 conversion, so staging is
// load->cvt->ds_store here (async copies cannot convert).
// ---------------------------------------------------------------------------
__global__ __launch_bounds__(256) void qkv_gemm(const float* __restrict__ x,
                                                const float* __restrict__ wqkv,
                                                _Float16* __restrict__ qkv) {
  __shared__ __align__(16) _Float16 Xs[128][40];
  __shared__ __align__(16) _Float16 Ws[64][40];   // staged transposed [n][k]
  const int tid = threadIdx.x;
  const int lane = tid & 31;
  const int wv = tid >> 5;
  const int mw = wv >> 1, nw = wv & 1;
  const int col0 = blockIdx.x * 64;
  const int row0 = blockIdx.y * 128;

  v8f acc[2][2];
#pragma unroll
  for (int i = 0; i < 2; ++i)
#pragma unroll
    for (int j = 0; j < 2; ++j) acc[i][j] = vzero8();

  for (int kk = 0; kk < 512; kk += 32) {
#pragma unroll
    for (int t = 0; t < 16; ++t) {            // 128x32 f32 -> f16 LDS
      int idx = tid + t * 256;
      int r = idx >> 5, c = idx & 31;
      Xs[r][c] = (_Float16)x[(size_t)(row0 + r) * 512 + kk + c];
    }
#pragma unroll
    for (int t = 0; t < 8; ++t) {             // 32x64 f32 -> f16 LDS (transposed)
      int idx = tid + t * 256;
      int k = idx >> 6, n = idx & 63;
      Ws[n][k] = (_Float16)wqkv[(size_t)(kk + k) * 1536 + col0 + n];
    }
    __syncthreads();
    v16h a0 = frag_ld(&Xs[mw * 32][0], 40);
    v16h a1 = frag_ld(&Xs[mw * 32 + 16][0], 40);
    v16h b0 = frag_ld(&Ws[nw * 32][0], 40);
    v16h b1 = frag_ld(&Ws[nw * 32 + 16][0], 40);
    acc[0][0] = wmma16(a0, b0, acc[0][0]);
    acc[0][1] = wmma16(a0, b1, acc[0][1]);
    acc[1][0] = wmma16(a1, b0, acc[1][0]);
    acc[1][1] = wmma16(a1, b1, acc[1][1]);
    __syncthreads();
  }

  const int hi = lane >> 4, lm = lane & 15;
#pragma unroll
  for (int mi = 0; mi < 2; ++mi)
#pragma unroll
    for (int ni = 0; ni < 2; ++ni) {
      int gcb = col0 + nw * 32 + ni * 16;     // 16-col frag lies in one head
      int which = gcb >> 9;
      int head = (gcb & 511) >> 6;
      int d = (gcb & 63) + lm;
#pragma unroll
      for (int r = 0; r < 8; ++r) {
        int gr = row0 + mw * 32 + mi * 16 + r + 8 * hi;
        int bb = gr >> 10, nn = gr & 1023;
        size_t off = ((((size_t)which * 32 + bb) * 8 + head) * 1024 + nn) * 64 + d;
        qkv[off] = (_Float16)acc[mi][ni][r];
      }
    }
}

// ---------------------------------------------------------------------------
// Kernel 2: local-window flash attention. One wave per (b, head, qtile).
// Q/K tiles are pure f16 copies -> GLOBAL_LOAD_ASYNC_TO_LDS_B128 (ASYNCcnt),
// V needs a transpose so it keeps the scalar staging path.
// ---------------------------------------------------------------------------
__global__ __launch_bounds__(32) void local_attn(const _Float16* __restrict__ qkv,
                                                 _Float16* __restrict__ attn) {
  __shared__ __align__(16) _Float16 Qs[16][72];
  __shared__ __align__(16) _Float16 Ks[16][72];
  __shared__ __align__(16) _Float16 Vs[64][40];  // transposed [d][key], keys 16..31 = 0
  __shared__ __align__(16) _Float16 Ps[16][40];  // [q][key], keys 16..31 = 0
  const int lane = threadIdx.x & 31;
  const int hi = lane >> 4, lm = lane & 15;
  const int qt = blockIdx.x, head = blockIdx.y, b = blockIdx.z;
  const int q0 = qt * 16;
  const int hq = qt >> 3;
  const int ctq = qt & 7;
  const int wq0 = ctq * 16;

  const size_t plane = (size_t)1024 * 64;
  const _Float16* Qg = qkv + (((size_t)0 * 32 + b) * 8 + head) * plane;
  const _Float16* Kg = qkv + (((size_t)1 * 32 + b) * 8 + head) * plane;
  const _Float16* Vg = qkv + (((size_t)2 * 32 + b) * 8 + head) * plane;

  // async-stage Q tile 16x64 (4 rounds x 32 lanes x 16B)
#pragma unroll
  for (int ch = lane; ch < 128; ch += 32) {
    int r = ch >> 3, c8 = (ch & 7) * 8;
    async_cp_b128(&Qg[(size_t)(q0 + r) * 64 + c8], &Qs[r][c8]);
  }
  for (int idx = lane; idx < 16 * 40; idx += 32) Ps[idx / 40][idx % 40] = (_Float16)0.f;
  for (int idx = lane; idx < 64 * 40; idx += 32) Vs[idx / 40][idx % 40] = (_Float16)0.f;
  wait_async0();
  __syncthreads();
  v16h qa0 = frag_ld(&Qs[0][0], 72);
  v16h qa1 = frag_ld(&Qs[0][32], 72);

  float mrun[8], lrun[8];
  v8f o[4];
#pragma unroll
  for (int r = 0; r < 8; ++r) { mrun[r] = NEG_BIG; lrun[r] = 0.f; }
#pragma unroll
  for (int dt = 0; dt < 4; ++dt) o[dt] = vzero8();

  for (int ik = hq - 3; ik <= hq + 3; ++ik) {
    if (ik < 0 || ik >= 8) continue;
    for (int c = ctq - 1; c <= ctq + 1; ++c) {
      if (c < 0 || c >= 8) continue;
      const int kb = ik * 128 + c * 16;
      __syncthreads();
#pragma unroll
      for (int ch = lane; ch < 128; ch += 32) {        // async K tile [key][d]
        int r = ch >> 3, c8 = (ch & 7) * 8;
        async_cp_b128(&Kg[(size_t)(kb + r) * 64 + c8], &Ks[r][c8]);
      }
      for (int idx = lane; idx < 1024; idx += 32) {    // V tile transposed [d][key]
        int key = idx >> 6, d = idx & 63;
        Vs[d][key] = Vg[(size_t)(kb + key) * 64 + d];
      }
      wait_async0();
      __syncthreads();

      v16h kb0 = frag_ld(&Ks[0][0], 72);
      v16h kb1 = frag_ld(&Ks[0][32], 72);
      v8f s = vzero8();
      s = wmma16(qa0, kb0, s);
      s = wmma16(qa1, kb1, s);

      const int wk = c * 16 + lm;                       // key w coordinate
#pragma unroll
      for (int r = 0; r < 8; ++r) {
        int Mq = r + 8 * hi;
        int dw = wk - (wq0 + Mq) + 5;                   // WK//2 = 5
        float sv = s[r] * SCALE_F;
        if (dw < 0 || dw >= 11) sv = NEG_BIG;
        float mt = sv;
        mt = fmaxf(mt, __shfl_xor(mt, 1));
        mt = fmaxf(mt, __shfl_xor(mt, 2));
        mt = fmaxf(mt, __shfl_xor(mt, 4));
        mt = fmaxf(mt, __shfl_xor(mt, 8));
        float mnew = fmaxf(fmaxf(mrun[r], mt), -1e30f);
        float p = __expf(sv - mnew);
        float rs = p;
        rs += __shfl_xor(rs, 1);
        rs += __shfl_xor(rs, 2);
        rs += __shfl_xor(rs, 4);
        rs += __shfl_xor(rs, 8);
        float corr = __expf(mrun[r] - mnew);
        lrun[r] = lrun[r] * corr + rs;
        mrun[r] = mnew;
#pragma unroll
        for (int dt = 0; dt < 4; ++dt) o[dt][r] *= corr;
        Ps[Mq][lm] = (_Float16)p;                       // D-layout -> LDS
      }
      __syncthreads();

      v16h pf = frag_ld(&Ps[0][0], 40);                 // back as A fragment
#pragma unroll
      for (int dt = 0; dt < 4; ++dt) {
        v16h vf = frag_ld(&Vs[dt * 16][0], 40);
        o[dt] = wmma16(pf, vf, o[dt]);
      }
    }
  }

#pragma unroll
  for (int dt = 0; dt < 4; ++dt)
#pragma unroll
    for (int r = 0; r < 8; ++r) {
      int Mq = r + 8 * hi;
      float val = o[dt][r] / lrun[r];
      attn[((size_t)b * 1024 + q0 + Mq) * 512 + head * 64 + dt * 16 + lm] = (_Float16)val;
    }
}

// ---------------------------------------------------------------------------
// Kernel 3: out(f32) = attn(f16)[32768x512] @ w_proj(f32)[512x512] + b_proj.
// A tile is a pure f16 copy -> async global->LDS; W needs f32->f16 convert.
// ---------------------------------------------------------------------------
__global__ __launch_bounds__(256) void proj_gemm(const _Float16* __restrict__ attn,
                                                 const float* __restrict__ wproj,
                                                 const float* __restrict__ bias,
                                                 float* __restrict__ out) {
  __shared__ __align__(16) _Float16 As[128][40];
  __shared__ __align__(16) _Float16 Ws[64][40];
  const int tid = threadIdx.x;
  const int lane = tid & 31;
  const int wv = tid >> 5;
  const int mw = wv >> 1, nw = wv & 1;
  const int col0 = blockIdx.x * 64;
  const int row0 = blockIdx.y * 128;

  v8f acc[2][2];
#pragma unroll
  for (int i = 0; i < 2; ++i)
#pragma unroll
    for (int j = 0; j < 2; ++j) acc[i][j] = vzero8();

  for (int kk = 0; kk < 512; kk += 32) {
#pragma unroll
    for (int t = 0; t < 2; ++t) {             // async 128x32 f16 tile, 16B/lane
      int idx = tid + t * 256;
      int r = idx >> 2, c8 = (idx & 3) * 8;
      async_cp_b128(&attn[(size_t)(row0 + r) * 512 + kk + c8], &As[r][c8]);
    }
#pragma unroll
    for (int t = 0; t < 8; ++t) {
      int idx = tid + t * 256;
      int k = idx >> 6, n = idx & 63;
      Ws[n][k] = (_Float16)wproj[(size_t)(kk + k) * 512 + col0 + n];
    }
    wait_async0();
    __syncthreads();
    v16h a0 = frag_ld(&As[mw * 32][0], 40);
    v16h a1 = frag_ld(&As[mw * 32 + 16][0], 40);
    v16h b0 = frag_ld(&Ws[nw * 32][0], 40);
    v16h b1 = frag_ld(&Ws[nw * 32 + 16][0], 40);
    acc[0][0] = wmma16(a0, b0, acc[0][0]);
    acc[0][1] = wmma16(a0, b1, acc[0][1]);
    acc[1][0] = wmma16(a1, b0, acc[1][0]);
    acc[1][1] = wmma16(a1, b1, acc[1][1]);
    __syncthreads();
  }

  const int hi = lane >> 4, lm = lane & 15;
#pragma unroll
  for (int mi = 0; mi < 2; ++mi)
#pragma unroll
    for (int ni = 0; ni < 2; ++ni) {
      int gc = col0 + nw * 32 + ni * 16 + lm;
      float bv = bias[gc];
#pragma unroll
      for (int r = 0; r < 8; ++r) {
        int gr = row0 + mw * 32 + mi * 16 + r + 8 * hi;
        out[(size_t)gr * 512 + gc] = acc[mi][ni][r] + bv;
      }
    }
}

extern "C" void kernel_launch(void* const* d_in, const int* in_sizes, int n_in,
                              void* d_out, int out_size, void* d_ws, size_t ws_size,
                              hipStream_t stream) {
  (void)in_sizes; (void)n_in; (void)out_size; (void)ws_size;
  const float* x     = (const float*)d_in[0];   // [32,1024,512]
  const float* wqkv  = (const float*)d_in[1];   // [512,1536]
  const float* wproj = (const float*)d_in[2];   // [512,512]
  const float* bias  = (const float*)d_in[3];   // [512]
  float* out = (float*)d_out;                   // [32,1024,512] f32

  // ws: qkv f16 [3][32][8][1024][64] = 96 MB, attn f16 [32][1024][512] = 32 MB
  _Float16* qkv  = (_Float16*)d_ws;
  _Float16* attn = qkv + (size_t)3 * 32 * 8 * 1024 * 64;

  qkv_gemm <<<dim3(24, 256), 256, 0, stream>>>(x, wqkv, qkv);
  local_attn<<<dim3(64, 8, 32),  32, 0, stream>>>(qkv, attn);
  proj_gemm <<<dim3(8, 256),  256, 0, stream>>>(attn, wproj, bias, out);
}
